// EncoderMoe_13048110645670
// MI455X (gfx1250) — matrix operand order repaired and checked by reading, not compiled, not run on hardware
//
#include <hip/hip_runtime.h>
#include <hip/hip_bf16.h>
#include <math.h>

// ---- problem constants (from reference) ----
#define BSZ   8
#define SEQ   256
#define TOK   2048        // B*S
#define HD    1024
#define NHEAD 16
#define DH    64
#define MLPD  4096
#define NEXP  8
#define QKVD  3072
#define NBH   128         // B*NHEAD

typedef __attribute__((ext_vector_type(16))) __bf16 v16bf;
typedef __attribute__((ext_vector_type(8)))  __bf16 v8bf;
typedef __attribute__((ext_vector_type(4)))  __bf16 v4bf;
typedef __attribute__((ext_vector_type(8)))  float  v8f;

struct Bf16Pair { v8bf lo; v8bf hi; };   // 32 bytes, no padding

// fp32 -> bf16 round-to-nearest-even (used only OUTSIDE inner loops)
__device__ __forceinline__ __bf16 f2bf(float f) {
  unsigned u = __builtin_bit_cast(unsigned, f);
  u += 0x7FFFu + ((u >> 16) & 1u);
  unsigned short us = (unsigned short)(u >> 16);
  return __builtin_bit_cast(__bf16, us);
}

__device__ __forceinline__ float gelu_exact(float v) {
  return 0.5f * v * (1.0f + erff(v * 0.70710678118654752f));
}

__device__ __forceinline__ v8f wmma_bf16(v16bf a, v16bf b, v8f c) {
  return __builtin_amdgcn_wmma_f32_16x16x32_bf16(false, a, false, b, (short)0, c,
                                                 false, false);
}

// A fragment 16x32: lane group g holds K run [g*16, g*16+16):
// two contiguous 16B loads at p and p+8 halves.
__device__ __forceinline__ v16bf frag_a(const __bf16* p) {
  Bf16Pair t;
  t.lo = *reinterpret_cast<const v8bf*>(p);
  t.hi = *reinterpret_cast<const v8bf*>(p + 8);
  return __builtin_bit_cast(v16bf, t);
}

// B fragment 32x16 (NT): lane group g holds K run [g*16, g*16+16):
// two contiguous 16B loads at p and p+8 halves.
__device__ __forceinline__ v16bf frag_b(const __bf16* p) {
  Bf16Pair t;
  t.lo = *reinterpret_cast<const v8bf*>(p);
  t.hi = *reinterpret_cast<const v8bf*>(p + 8);
  return __builtin_bit_cast(v16bf, t);
}

// B fragment (NN: V stored [K,N] row-major) — strided gather, no conversion
__device__ __forceinline__ v16bf frag_b_nn(const __bf16* Vb, int ld, int n, int k0,
                                           int g) {
  const __bf16* p = Vb + (size_t)(k0 + g * 16) * ld + n;
  v16bf b;
#pragma unroll
  for (int h = 0; h < 16; ++h) b[h] = p[(size_t)h * ld];
  return b;
}

// ---------------- generic WMMA GEMM ----------------
//  acc = X[M,K](bf16) * W[N,K](bf16)^T ; v = act(acc + bias) * row_scale
//  Yf (f32): accum? Yf+=v : Yf=v     Yh (bf16): Yh = v
// block = 256 thr = 8 waves; wave tile 64x32 (4 m-subtiles x 2 n-subtiles)
// K-loop steps by 64 with two fragment sets so half the loads hide under WMMAs.
__global__ __launch_bounds__(256) void gemm_bf16_nt(
    float* __restrict__ Yf, __bf16* __restrict__ Yh,
    const __bf16* __restrict__ X, const __bf16* __restrict__ W,
    const float* __restrict__ bias, const float* __restrict__ row_scale, int rs_stride,
    int M, int N, int K, int act, int accum) {
  const int lane = threadIdx.x & 31;
  const int wave = threadIdx.x >> 5;
  const int nl = lane & 15;
  const int g = lane >> 4;
  const int n0 = blockIdx.x * 256 + wave * 32;
  const int m0 = blockIdx.y * 64;

  v8f acc[4][2];
#pragma unroll
  for (int i = 0; i < 4; ++i)
#pragma unroll
    for (int j = 0; j < 2; ++j) acc[i][j] = (v8f){0.f,0.f,0.f,0.f,0.f,0.f,0.f,0.f};

  size_t offA[4], offB[2];
#pragma unroll
  for (int i = 0; i < 4; ++i) offA[i] = (size_t)(m0 + 16 * i + nl) * K + g * 16;
#pragma unroll
  for (int j = 0; j < 2; ++j) offB[j] = (size_t)(n0 + 16 * j + nl) * K + g * 16;

  for (int k0 = 0; k0 < K; k0 += 64) {
    if (k0 + 128 < K)
      __builtin_prefetch(W + offB[0] + k0 + 128, 0, 3);  // WGP-scope prefetch
    v16bf a0[4], a1[4], b0[2], b1[2];
#pragma unroll
    for (int i = 0; i < 4; ++i) {
      a0[i] = frag_a(X + offA[i] + k0);
      a1[i] = frag_a(X + offA[i] + k0 + 32);
    }
#pragma unroll
    for (int j = 0; j < 2; ++j) {
      b0[j] = frag_b(W + offB[j] + k0);
      b1[j] = frag_b(W + offB[j] + k0 + 32);
    }
#pragma unroll
    for (int i = 0; i < 4; ++i)
#pragma unroll
      for (int j = 0; j < 2; ++j) acc[i][j] = wmma_bf16(a0[i], b0[j], acc[i][j]);
#pragma unroll
    for (int i = 0; i < 4; ++i)
#pragma unroll
      for (int j = 0; j < 2; ++j) acc[i][j] = wmma_bf16(a1[i], b1[j], acc[i][j]);
  }

#pragma unroll
  for (int i = 0; i < 4; ++i)
#pragma unroll
    for (int j = 0; j < 2; ++j) {
      const int n = n0 + 16 * j + nl;
      const float bv = bias ? bias[n] : 0.f;
#pragma unroll
      for (int r = 0; r < 8; ++r) {
        const int m = m0 + 16 * i + r + 8 * g;
        float v = acc[i][j][r] + bv;
        if (act) v = gelu_exact(v);
        if (row_scale) v *= row_scale[(size_t)m * rs_stride];
        const size_t idx = (size_t)m * N + n;
        if (Yf) { if (accum) Yf[idx] += v; else Yf[idx] = v; }
        if (Yh) Yh[idx] = f2bf(v);
      }
    }
}

// ---------------- attention: S = Q*K^T * 1/sqrt(dh), batched over (b,h) ----------
// grid (SEQ/64, NBH), block 256 (8 waves, each 64x32 of the 256x256 score tile)
__global__ __launch_bounds__(256) void attn_scores_kernel(float* __restrict__ S,
                                                          const __bf16* __restrict__ qkv) {
  const int lane = threadIdx.x & 31, wave = threadIdx.x >> 5;
  const int nl = lane & 15, g = lane >> 4;
  const int bh = blockIdx.y, b = bh >> 4, h = bh & 15;
  const size_t qbase = (size_t)b * SEQ * QKVD + h * DH;
  float* Sp = S + (size_t)bh * SEQ * SEQ;
  const int m0 = blockIdx.x * 64;
  const int n0 = wave * 32;

  v8f acc[4][2];
#pragma unroll
  for (int i = 0; i < 4; ++i)
#pragma unroll
    for (int j = 0; j < 2; ++j) acc[i][j] = (v8f){0.f,0.f,0.f,0.f,0.f,0.f,0.f,0.f};

  size_t offA[4], offB[2];
#pragma unroll
  for (int i = 0; i < 4; ++i)
    offA[i] = qbase + (size_t)(m0 + 16 * i + nl) * QKVD + g * 16;         // Q
#pragma unroll
  for (int j = 0; j < 2; ++j)
    offB[j] = qbase + HD + (size_t)(n0 + 16 * j + nl) * QKVD + g * 16;    // K

  // K-dim = 64: both halves fully unrolled
  v16bf a0[4], a1[4], b0[2], b1[2];
#pragma unroll
  for (int i = 0; i < 4; ++i) {
    a0[i] = frag_a(qkv + offA[i]);
    a1[i] = frag_a(qkv + offA[i] + 32);
  }
#pragma unroll
  for (int j = 0; j < 2; ++j) {
    b0[j] = frag_b(qkv + offB[j]);
    b1[j] = frag_b(qkv + offB[j] + 32);
  }
#pragma unroll
  for (int i = 0; i < 4; ++i)
#pragma unroll
    for (int j = 0; j < 2; ++j) acc[i][j] = wmma_bf16(a0[i], b0[j], acc[i][j]);
#pragma unroll
  for (int i = 0; i < 4; ++i)
#pragma unroll
    for (int j = 0; j < 2; ++j) acc[i][j] = wmma_bf16(a1[i], b1[j], acc[i][j]);

  const float scale = 0.125f;  // 1/sqrt(64)
#pragma unroll
  for (int i = 0; i < 4; ++i)
#pragma unroll
    for (int j = 0; j < 2; ++j)
#pragma unroll
      for (int r = 0; r < 8; ++r) {
        const int m = m0 + 16 * i + r + 8 * g;
        const int n = n0 + 16 * j + nl;
        Sp[(size_t)m * SEQ + n] = acc[i][j][r] * scale;
      }
}

// ---------------- O = P * V batched; block = (b,h); 8 waves each 32 rows x 64 cols
__global__ __launch_bounds__(256) void attn_out_kernel(__bf16* __restrict__ O,
                                                       const __bf16* __restrict__ P,
                                                       const __bf16* __restrict__ qkv) {
  const int lane = threadIdx.x & 31, wave = threadIdx.x >> 5;
  const int nl = lane & 15, g = lane >> 4;
  const int bh = blockIdx.x, b = bh >> 4, h = bh & 15;
  const size_t pbase = (size_t)bh * SEQ * SEQ;
  const __bf16* V = qkv + (size_t)b * SEQ * QKVD + 2 * HD + h * DH;
  const int m0 = wave * 32;

  v8f acc[2][4];
#pragma unroll
  for (int i = 0; i < 2; ++i)
#pragma unroll
    for (int j = 0; j < 4; ++j) acc[i][j] = (v8f){0.f,0.f,0.f,0.f,0.f,0.f,0.f,0.f};

  size_t offA[2];
#pragma unroll
  for (int i = 0; i < 2; ++i) offA[i] = pbase + (size_t)(m0 + 16 * i + nl) * SEQ + g * 16;

  for (int k0 = 0; k0 < SEQ; k0 += 64) {
    v16bf a0[2], a1[2], b0[4], b1[4];
#pragma unroll
    for (int i = 0; i < 2; ++i) {
      a0[i] = frag_a(P + offA[i] + k0);
      a1[i] = frag_a(P + offA[i] + k0 + 32);
    }
#pragma unroll
    for (int j = 0; j < 4; ++j) {
      b0[j] = frag_b_nn(V, QKVD, 16 * j + nl, k0, g);
      b1[j] = frag_b_nn(V, QKVD, 16 * j + nl, k0 + 32, g);
    }
#pragma unroll
    for (int i = 0; i < 2; ++i)
#pragma unroll
      for (int j = 0; j < 4; ++j) acc[i][j] = wmma_bf16(a0[i], b0[j], acc[i][j]);
#pragma unroll
    for (int i = 0; i < 2; ++i)
#pragma unroll
      for (int j = 0; j < 4; ++j) acc[i][j] = wmma_bf16(a1[i], b1[j], acc[i][j]);
  }

#pragma unroll
  for (int i = 0; i < 2; ++i)
#pragma unroll
    for (int j = 0; j < 4; ++j)
#pragma unroll
      for (int r = 0; r < 8; ++r) {
        const int q = m0 + 16 * i + r + 8 * g;
        const int d = 16 * j + nl;
        O[(size_t)(b * SEQ + q) * HD + h * DH + d] = f2bf(acc[i][j][r]);
      }
}

// ---------------- pointwise / reduction kernels ----------------
__global__ void f32_to_bf16_kernel(__bf16* __restrict__ o, const float* __restrict__ in,
                                   int n4) {
  const int i = blockIdx.x * blockDim.x + threadIdx.x;
  if (i >= n4) return;
  const float4 f = reinterpret_cast<const float4*>(in)[i];
  v4bf r;
  r[0] = f2bf(f.x); r[1] = f2bf(f.y); r[2] = f2bf(f.z); r[3] = f2bf(f.w);
  reinterpret_cast<v4bf*>(o)[i] = r;
}

__global__ void add_pos_kernel(float* __restrict__ x, const float* __restrict__ xin,
                               const float* __restrict__ pos) {
  const size_t idx = (size_t)blockIdx.x * blockDim.x + threadIdx.x;
  const int col = (int)(idx & (HD - 1));
  const int row = (int)(idx >> 10);
  const int s = row & (SEQ - 1);
  x[idx] = xin[idx] + pos[(size_t)s * HD + col];
}

__device__ __forceinline__ float block_sum(float v, float* sm) {
  const int tid = threadIdx.x;
  sm[tid] = v; __syncthreads();
  for (int s = 128; s > 0; s >>= 1) {
    if (tid < s) sm[tid] += sm[tid + s];
    __syncthreads();
  }
  const float r = sm[0]; __syncthreads();
  return r;
}

// LN -> f32 out (optional) + bf16 out (optional)
__global__ __launch_bounds__(256) void layernorm_kernel(float* __restrict__ outf,
                                                        __bf16* __restrict__ outh,
                                                        const float* __restrict__ in,
                                                        const float* __restrict__ w,
                                                        const float* __restrict__ b) {
  __shared__ float sm[256];
  const int row = blockIdx.x, tid = threadIdx.x;
  const float* p = in + (size_t)row * HD;
  float s = 0.f;
  for (int c = tid; c < HD; c += 256) s += p[c];
  const float mu = block_sum(s, sm) * (1.0f / HD);
  float vs = 0.f;
  for (int c = tid; c < HD; c += 256) { const float d = p[c] - mu; vs += d * d; }
  const float var = block_sum(vs, sm) * (1.0f / HD);
  const float inv = rsqrtf(var + 1e-5f);
  for (int c = tid; c < HD; c += 256) {
    const float v = (p[c] - mu) * inv * w[c] + b[c];
    if (outf) outf[(size_t)row * HD + c] = v;
    if (outh) outh[(size_t)row * HD + c] = f2bf(v);
  }
}

// rowwise softmax over 256, f32 in -> bf16 probabilities out
__global__ __launch_bounds__(256) void softmax256_kernel(__bf16* __restrict__ P,
                                                         const float* __restrict__ S) {
  __shared__ float sm[256];
  const int tid = threadIdx.x;
  const float* p = S + (size_t)blockIdx.x * SEQ;
  const float v = p[tid];
  sm[tid] = v; __syncthreads();
  for (int s = 128; s > 0; s >>= 1) {
    if (tid < s) sm[tid] = fmaxf(sm[tid], sm[tid + s]);
    __syncthreads();
  }
  const float mx = sm[0]; __syncthreads();
  const float e = expf(v - mx);
  const float sum = block_sum(e, sm);
  P[(size_t)blockIdx.x * SEQ + tid] = f2bf(e / sum);
}

__global__ __launch_bounds__(256) void router_kernel(float* __restrict__ logits,
                                                     const float* __restrict__ X,
                                                     const float* __restrict__ Wr) {
  __shared__ float sm[256];
  const int row = blockIdx.x, tid = threadIdx.x;
  const float* xr = X + (size_t)row * HD;
  float accu[NEXP];
#pragma unroll
  for (int e = 0; e < NEXP; ++e) accu[e] = 0.f;
  for (int hh = tid; hh < HD; hh += 256) {
    const float xv = xr[hh];
    const float* wrow = Wr + (size_t)hh * NEXP;
#pragma unroll
    for (int e = 0; e < NEXP; ++e) accu[e] += xv * wrow[e];
  }
  for (int e = 0; e < NEXP; ++e) {
    const float r = block_sum(accu[e], sm);
    if (tid == 0) logits[(size_t)row * NEXP + e] = r;
  }
}

__global__ void gates_kernel(float* __restrict__ gates, float* __restrict__ probs,
                             const float* __restrict__ logits) {
  const int row = blockIdx.x * blockDim.x + threadIdx.x;
  if (row >= TOK) return;
  const float* l = logits + (size_t)row * NEXP;
  float mx = -1e30f;
#pragma unroll
  for (int e = 0; e < NEXP; ++e) mx = fmaxf(mx, l[e]);
  float p[NEXP], sum = 0.f;
#pragma unroll
  for (int e = 0; e < NEXP; ++e) { p[e] = expf(l[e] - mx); sum += p[e]; }
  float m1 = -1.f, m2 = -1.f;
#pragma unroll
  for (int e = 0; e < NEXP; ++e) {
    p[e] /= sum;
    if (p[e] > m1) { m2 = m1; m1 = p[e]; } else if (p[e] > m2) { m2 = p[e]; }
  }
#pragma unroll
  for (int e = 0; e < NEXP; ++e) {
    probs[(size_t)row * NEXP + e] = p[e];
    gates[(size_t)row * NEXP + e] = (p[e] >= m2) ? p[e] : 0.f;
  }
}

// aux += (std(imp)/(mean(imp)+1e-9))^2 ; imp[e] = sum over tokens of probs[:,e]
__global__ __launch_bounds__(256) void aux_kernel(float* __restrict__ aux_out,
                                                  const float* __restrict__ probs) {
  __shared__ float sm[256];
  __shared__ float imp[NEXP];
  const int tid = threadIdx.x;
  float part[NEXP];
#pragma unroll
  for (int e = 0; e < NEXP; ++e) part[e] = 0.f;
  for (int row = tid; row < TOK; row += 256) {
#pragma unroll
    for (int e = 0; e < NEXP; ++e) part[e] += probs[(size_t)row * NEXP + e];
  }
  for (int e = 0; e < NEXP; ++e) {
    const float r = block_sum(part[e], sm);
    if (tid == 0) imp[e] = r;
  }
  __syncthreads();
  if (tid == 0) {
    float mean = 0.f;
    for (int e = 0; e < NEXP; ++e) mean += imp[e];
    mean *= (1.0f / NEXP);
    float var = 0.f;
    for (int e = 0; e < NEXP; ++e) { const float d = imp[e] - mean; var += d * d; }
    var *= (1.0f / NEXP);
    const float cv = sqrtf(var) / (mean + 1e-9f);
    *aux_out += cv * cv;
  }
}

__global__ void zero_kernel(float* p) { *p = 0.f; }

// ---------------------------------------------------------------------------
extern "C" void kernel_launch(void* const* d_in, const int* in_sizes, int n_in,
                              void* d_out, int out_size, void* d_ws, size_t ws_size,
                              hipStream_t stream) {
  const float* in_x       = (const float*)d_in[0];
  const float* pos_emb    = (const float*)d_in[1];
  const float* ln1_w      = (const float*)d_in[2];
  const float* ln1_b      = (const float*)d_in[3];
  const float* attn_in_w  = (const float*)d_in[4];
  const float* attn_in_b  = (const float*)d_in[5];
  const float* attn_out_w = (const float*)d_in[6];
  const float* attn_out_b = (const float*)d_in[7];
  const float* ln2_w      = (const float*)d_in[8];
  const float* ln2_b      = (const float*)d_in[9];
  const float* dfc1_w     = (const float*)d_in[10];
  const float* dfc1_b     = (const float*)d_in[11];
  const float* dfc2_w     = (const float*)d_in[12];
  const float* dfc2_b     = (const float*)d_in[13];
  const float* mfc1_w     = (const float*)d_in[14];
  const float* mfc1_b     = (const float*)d_in[15];
  const float* mfc2_w     = (const float*)d_in[16];
  const float* mfc2_b     = (const float*)d_in[17];
  const float* router_w   = (const float*)d_in[18];
  const float* fln_w      = (const float*)d_in[19];
  const float* fln_b      = (const float*)d_in[20];
  float* out = (float*)d_out;

  // ---- workspace carve (256B aligned slabs) ----
  char* base = (char*)d_ws;
  auto carve = [&](size_t bytes) {
    char* p = base;
    base += (bytes + 255) & ~(size_t)255;
    return p;
  };
  float*  x      = (float*)carve((size_t)TOK * HD * 4);          // residual stream
  float*  lnf    = (float*)carve((size_t)TOK * HD * 4);          // LN out f32
  float*  sc     = (float*)carve((size_t)NBH * SEQ * SEQ * 4);   // raw scores
  float*  logits = (float*)carve((size_t)TOK * NEXP * 4);
  float*  probs  = (float*)carve((size_t)TOK * NEXP * 4);
  float*  gates  = (float*)carve((size_t)TOK * NEXP * 4);
  __bf16* lnh    = (__bf16*)carve((size_t)TOK * HD * 2);         // LN out bf16
  __bf16* qkv    = (__bf16*)carve((size_t)TOK * QKVD * 2);       // QKV bf16
  __bf16* pbf    = (__bf16*)carve((size_t)NBH * SEQ * SEQ * 2);  // softmax probs
  __bf16* ob     = (__bf16*)carve((size_t)TOK * HD * 2);         // attn out bf16
  __bf16* h1     = (__bf16*)carve((size_t)TOK * MLPD * 2);       // fc1 out bf16
  __bf16* wbf    = (__bf16*)carve((size_t)MLPD * HD * 2);        // weight staging

  auto cvt = [&](__bf16* dst, const float* src, size_t n) {
    const int n4 = (int)(n / 4);
    f32_to_bf16_kernel<<<(n4 + 255) / 256, 256, 0, stream>>>(dst, src, n4);
  };

  zero_kernel<<<1, 1, 0, stream>>>(out + (size_t)TOK * HD);
  add_pos_kernel<<<(TOK * HD) / 256, 256, 0, stream>>>(x, in_x, pos_emb);

  int di = 0, mi = 0;
  for (int l = 0; l < 4; ++l) {
    // --- attention block ---
    layernorm_kernel<<<TOK, 256, 0, stream>>>(nullptr, lnh, x,
                                              ln1_w + l * HD, ln1_b + l * HD);
    cvt(wbf, attn_in_w + (size_t)l * QKVD * HD, (size_t)QKVD * HD);
    gemm_bf16_nt<<<dim3(QKVD / 256, TOK / 64), 256, 0, stream>>>(
        nullptr, qkv, lnh, wbf, attn_in_b + (size_t)l * QKVD,
        nullptr, 0, TOK, QKVD, HD, 0, 0);
    attn_scores_kernel<<<dim3(SEQ / 64, NBH), 256, 0, stream>>>(sc, qkv);
    softmax256_kernel<<<NBH * SEQ, 256, 0, stream>>>(pbf, sc);
    attn_out_kernel<<<NBH, 256, 0, stream>>>(ob, pbf, qkv);
    cvt(wbf, attn_out_w + (size_t)l * HD * HD, (size_t)HD * HD);
    gemm_bf16_nt<<<dim3(HD / 256, TOK / 64), 256, 0, stream>>>(
        x, nullptr, ob, wbf, attn_out_b + (size_t)l * HD,
        nullptr, 0, TOK, HD, HD, 0, 1);  // x += proj(o)

    // --- FFN block ---
    layernorm_kernel<<<TOK, 256, 0, stream>>>(lnf, lnh, x,
                                              ln2_w + l * HD, ln2_b + l * HD);
    if (l == 1 || l == 3) {  // MoE layer
      router_kernel<<<TOK, 256, 0, stream>>>(logits, lnf,
                                             router_w + (size_t)mi * HD * NEXP);
      gates_kernel<<<TOK / 256, 256, 0, stream>>>(gates, probs, logits);
      for (int e = 0; e < NEXP; ++e) {
        const float* w1 = mfc1_w + ((size_t)mi * NEXP + e) * MLPD * HD;
        const float* b1 = mfc1_b + ((size_t)mi * NEXP + e) * MLPD;
        const float* w2 = mfc2_w + ((size_t)mi * NEXP + e) * HD * MLPD;
        const float* b2 = mfc2_b + ((size_t)mi * NEXP + e) * HD;
        cvt(wbf, w1, (size_t)MLPD * HD);
        gemm_bf16_nt<<<dim3(MLPD / 256, TOK / 64), 256, 0, stream>>>(
            nullptr, h1, lnh, wbf, b1, nullptr, 0, TOK, MLPD, HD, 1, 0);
        cvt(wbf, w2, (size_t)HD * MLPD);
        gemm_bf16_nt<<<dim3(HD / 256, TOK / 64), 256, 0, stream>>>(
            x, nullptr, h1, wbf, b2, gates + e, NEXP, TOK, HD, MLPD, 0, 1);
      }
      aux_kernel<<<1, 256, 0, stream>>>(out + (size_t)TOK * HD, probs);
      ++mi;
    } else {  // dense MLP
      cvt(wbf, dfc1_w + (size_t)di * MLPD * HD, (size_t)MLPD * HD);
      gemm_bf16_nt<<<dim3(MLPD / 256, TOK / 64), 256, 0, stream>>>(
          nullptr, h1, lnh, wbf, dfc1_b + (size_t)di * MLPD,
          nullptr, 0, TOK, MLPD, HD, 1, 0);
      cvt(wbf, dfc2_w + (size_t)di * HD * MLPD, (size_t)HD * MLPD);
      gemm_bf16_nt<<<dim3(HD / 256, TOK / 64), 256, 0, stream>>>(
          x, nullptr, h1, wbf, dfc2_b + (size_t)di * HD,
          nullptr, 0, TOK, HD, MLPD, 0, 1);
      ++di;
    }
  }
  layernorm_kernel<<<TOK, 256, 0, stream>>>(out, nullptr, x, fln_w, fln_b);
}
